// AtomicEnsemble_33981781246533
// MI455X (gfx1250) — matrix operand order repaired and checked by reading, not compiled, not run on hardware
//
#include <hip/hip_runtime.h>
#include <hip/hip_bf16.h>

// ---------------------------------------------------------------------------
// ANI-style species-expert MLP ensemble on MI455X (gfx1250).
// Strategy: counting-scatter atoms into species buckets, then bucketed
// bf16 WMMA GEMM MLP (384->160->128->96->1, CELU) per species, scatter
// per-atom energies, reduce per molecule. ~12.4 GFLOP vs 49 GFLOP dense.
// ---------------------------------------------------------------------------

typedef __bf16 bf16_t;
typedef __attribute__((ext_vector_type(16))) __bf16 v16bf;
typedef __attribute__((ext_vector_type(8)))  __bf16 v8bf;
typedef __attribute__((ext_vector_type(8)))  float  v8f;

#define N_ATOMS 65536
#define BMOL    2048
#define APM     32
#define DAEV    384
#define S_SPEC  4
#define F1      160
#define F2      128
#define F3      96
#define F4P     16          // layer-4 output padded 1 -> 16
#define CELU_ALPHA 0.1f

#define TILE_M  48          // atoms per workgroup (3 WMMA M-tiles)
#define MT      (TILE_M/16) // 3
#define TILES   ((N_ATOMS + TILE_M - 1) / TILE_M)   // 1366

// swizzled weight sizes per species, in bf16 elements: [kblk][n][32]
#define L1_SW (12*F1*32)    // 61440  (K=384 -> 12 kblks)
#define L2_SW (5*F2*32)     // 20480  (K=160 -> 5)
#define L3_SW (4*F3*32)     // 12288  (K=128 -> 4)
#define L4_SW (3*F4P*32)    // 1536   (K=96  -> 3)
#define SPEC_SW (L1_SW + L2_SW + L3_SW + L4_SW)     // 95744

__device__ __forceinline__ bf16_t f2bf(float f) { return (bf16_t)f; }

// ---------------------------------------------------------------------------
// Kernel 1: swizzle f32 weights into bf16 fragment-friendly layout.
// Layout per (species, layer): sw[(kblk*Np + n)*32 + j], j = k - kblk*32.
// Lane (n16, hi) of a WMMA B-fragment then reads 16 contiguous halves at
// (kblk*Np + n)*32 + hi*16  (K = hi*16 + e), the same contiguous k order the A loads use.
// ---------------------------------------------------------------------------
__global__ void __launch_bounds__(256)
prep_weights_kernel(const float* __restrict__ W1, const float* __restrict__ W2,
                    const float* __restrict__ W3, const float* __restrict__ W4,
                    bf16_t* __restrict__ wsw)
{
    int gid = blockIdx.x * 256 + threadIdx.x;
    if (gid >= S_SPEC * SPEC_SW) return;
    int s = gid / SPEC_SW;
    int r = gid % SPEC_SW;

    const float* W; int Din, Dout, Np, idx;
    if (r < L1_SW)                        { W = W1; Din = DAEV; Dout = F1; Np = F1;  idx = r; }
    else if (r < L1_SW + L2_SW)           { W = W2; Din = F1;   Dout = F2; Np = F2;  idx = r - L1_SW; }
    else if (r < L1_SW + L2_SW + L3_SW)   { W = W3; Din = F2;   Dout = F3; Np = F3;  idx = r - (L1_SW + L2_SW); }
    else                                  { W = W4; Din = F3;   Dout = 1;  Np = F4P; idx = r - (L1_SW + L2_SW + L3_SW); }

    int kblk = idx / (Np * 32);
    int rr   = idx % (Np * 32);
    int n    = rr / 32;
    int j    = rr % 32;
    int k    = kblk * 32 + j;
    float v = 0.0f;
    if (k < Din && n < Dout) v = W[((long)s * Din + k) * Dout + n];
    wsw[gid] = f2bf(v);
}

// ---------------------------------------------------------------------------
// Kernel 2: counting scatter of atoms into species buckets.
// ---------------------------------------------------------------------------
__global__ void __launch_bounds__(256)
bucket_kernel(const int* __restrict__ species, int* __restrict__ counts,
              int* __restrict__ perm)
{
    int n = blockIdx.x * 256 + threadIdx.x;
    if (n >= N_ATOMS) return;
    int s = species[n];
    int idx = atomicAdd(&counts[s], 1);
    perm[s * N_ATOMS + idx] = n;
}

// ---------------------------------------------------------------------------
// One MLP layer on a 48-atom tile: LDS(bf16) x swizzled-weights -> LDS(bf16).
// A fragment (16x32 bf16): lane m=lane%16, halves K = hi*16+0..15 (contiguous)
// B fragment (32x16 bf16): lane n=lane%16, halves K = hi*16+0..15 (contiguous)
// D (16x16 f32): reg r -> M = r + 8*hi, N = lane%16
// ---------------------------------------------------------------------------
template<int DIN, int DOUTP, int DOUT>
__device__ __forceinline__ void layer_gemm(const bf16_t* __restrict__ lin,
                                           const bf16_t* __restrict__ wsw,
                                           const float*  __restrict__ bias,
                                           bf16_t* __restrict__ lout)
{
    const int wave = threadIdx.x >> 5;
    const int lane = threadIdx.x & 31;
    const int n16  = lane & 15;
    const int hi   = lane >> 4;
    constexpr int NT = DOUTP / 16;
    constexpr int KB = DIN / 32;

    for (int t = wave; t < MT * NT; t += 8) {
        const int mt = t / NT;
        const int nt = t % NT;
        v8f acc = {};
#pragma unroll
        for (int kb = 0; kb < KB; ++kb) {
            const bf16_t* ar = lin + (mt * 16 + n16) * DIN + kb * 32 + hi * 16;
            v8bf alo = *(const v8bf*)ar;          // K = hi*16 .. hi*16+7
            v8bf ahi = *(const v8bf*)(ar + 8);    // K = hi*16+8 ..
            v16bf a;
#pragma unroll
            for (int i = 0; i < 8; ++i) { a[i] = alo[i]; a[8 + i] = ahi[i]; }
            v16bf b = *(const v16bf*)(wsw + (kb * DOUTP + nt * 16 + n16) * 32 + hi * 16);
            acc = __builtin_amdgcn_wmma_f32_16x16x32_bf16(
                false, a, false, b, (short)0, acc, false, false);
        }
        const int nidx = nt * 16 + n16;
        const float bn = (nidx < DOUT) ? bias[nidx] : 0.0f;
#pragma unroll
        for (int r = 0; r < 8; ++r) {
            float v = acc[r] + bn;
            v = (v > 0.0f) ? v : CELU_ALPHA * (__expf(v * (1.0f / CELU_ALPHA)) - 1.0f);
            lout[(mt * 16 + r + hi * 8) * DOUTP + nidx] = f2bf(v);
        }
    }
}

// ---------------------------------------------------------------------------
// Kernel 3: full MLP on one (species, atom-tile). 256 threads = 8 wave32.
// ---------------------------------------------------------------------------
__global__ void __launch_bounds__(256)
mlp_kernel(const float* __restrict__ aev,
           const int*   __restrict__ perm,
           const int*   __restrict__ counts,
           const bf16_t* __restrict__ wsw,
           const float* __restrict__ b1, const float* __restrict__ b2,
           const float* __restrict__ b3, const float* __restrict__ b4,
           float* __restrict__ atom_e)
{
    __shared__ bf16_t buf0[TILE_M * DAEV];   // 48x384  (also holds h2 48x128)
    __shared__ bf16_t buf1[TILE_M * F1];     // 48x160  (also holds h3 48x96)

    const int s    = blockIdx.x / TILES;
    const int tile = blockIdx.x % TILES;
    const int cnt  = counts[s];
    const int row0 = tile * TILE_M;
    if (row0 >= cnt) return;
    const int* permS = perm + s * N_ATOMS;

    // Gather AEV rows for this species tile -> LDS bf16 (zero-pad tail rows).
    for (int i = threadIdx.x; i < TILE_M * DAEV; i += 256) {
        const int r = i / DAEV, c = i % DAEV;
        float v = 0.0f;
        if (row0 + r < cnt) v = aev[(long)permS[row0 + r] * DAEV + c];
        buf0[i] = f2bf(v);
    }
    __syncthreads();

    const bf16_t* ws = wsw + s * SPEC_SW;
    layer_gemm<DAEV, F1, F1>(buf0, ws,                         b1 + s * F1, buf1);
    __syncthreads();
    layer_gemm<F1,   F2, F2>(buf1, ws + L1_SW,                 b2 + s * F2, buf0);
    __syncthreads();
    layer_gemm<F2,   F3, F3>(buf0, ws + L1_SW + L2_SW,         b3 + s * F3, buf1);
    __syncthreads();

    // Layer 4: 96 -> 1 (padded to 16); waves 0..2 each take one M-tile.
    const int wave = threadIdx.x >> 5;
    const int lane = threadIdx.x & 31;
    const int n16  = lane & 15;
    const int hi   = lane >> 4;
    if (wave < MT) {
        const int mt = wave;
        const bf16_t* w4 = ws + L1_SW + L2_SW + L3_SW;
        v8f acc = {};
#pragma unroll
        for (int kb = 0; kb < 3; ++kb) {
            const bf16_t* ar = buf1 + (mt * 16 + n16) * F3 + kb * 32 + hi * 16;
            v8bf alo = *(const v8bf*)ar;
            v8bf ahi = *(const v8bf*)(ar + 8);
            v16bf a;
#pragma unroll
            for (int i = 0; i < 8; ++i) { a[i] = alo[i]; a[8 + i] = ahi[i]; }
            v16bf b = *(const v16bf*)(w4 + (kb * F4P + n16) * 32 + hi * 16);
            acc = __builtin_amdgcn_wmma_f32_16x16x32_bf16(
                false, a, false, b, (short)0, acc, false, false);
        }
        if (n16 == 0) {
            const float bb = b4[s];
#pragma unroll
            for (int r = 0; r < 8; ++r) {
                const int m = mt * 16 + r + hi * 8;
                if (row0 + m < cnt)
                    atom_e[permS[row0 + m]] = acc[r] + bb;   // no activation on L4
            }
        }
    }
}

// ---------------------------------------------------------------------------
// Kernel 4: deterministic per-molecule reduction (32 atoms each).
// ---------------------------------------------------------------------------
__global__ void __launch_bounds__(256)
reduce_kernel(const float* __restrict__ atom_e, float* __restrict__ out)
{
    int b = blockIdx.x * 256 + threadIdx.x;
    if (b >= BMOL) return;
    float s = 0.0f;
#pragma unroll
    for (int j = 0; j < APM; ++j) s += atom_e[b * APM + j];
    out[b] = s;
}

// ---------------------------------------------------------------------------
extern "C" void kernel_launch(void* const* d_in, const int* in_sizes, int n_in,
                              void* d_out, int out_size, void* d_ws, size_t ws_size,
                              hipStream_t stream) {
    const int*   species = (const int*)  d_in[0];
    const float* aev     = (const float*)d_in[1];
    const float* W1      = (const float*)d_in[2];
    const float* b1      = (const float*)d_in[3];
    const float* W2      = (const float*)d_in[4];
    const float* b2      = (const float*)d_in[5];
    const float* W3      = (const float*)d_in[6];
    const float* b3      = (const float*)d_in[7];
    const float* W4      = (const float*)d_in[8];
    const float* b4      = (const float*)d_in[9];
    float* out = (float*)d_out;

    // Workspace layout (~2.1 MB total):
    char*   ws      = (char*)d_ws;
    int*    counts  = (int*)ws;                                     // 256 B slot
    int*    perm    = (int*)(ws + 256);                             // 1 MB
    float*  atom_e  = (float*)(ws + 256 + sizeof(int) * S_SPEC * N_ATOMS);
    bf16_t* wsw     = (bf16_t*)(ws + 256 + sizeof(int) * S_SPEC * N_ATOMS
                                        + sizeof(float) * N_ATOMS);

    hipMemsetAsync(counts, 0, 256, stream);

    prep_weights_kernel<<<(S_SPEC * SPEC_SW + 255) / 256, 256, 0, stream>>>(
        W1, W2, W3, W4, wsw);

    bucket_kernel<<<(N_ATOMS + 255) / 256, 256, 0, stream>>>(species, counts, perm);

    mlp_kernel<<<S_SPEC * TILES, 256, 0, stream>>>(
        aev, perm, counts, wsw, b1, b2, b3, b4, atom_e);

    reduce_kernel<<<(BMOL + 255) / 256, 256, 0, stream>>>(atom_e, out);
}
